// SpatialGaugeCovariantBlocker_42004780155664
// MI455X (gfx1250) — matrix operand order repaired and checked by reading, not compiled, not run on hardware
//
#include <hip/hip_runtime.h>
#include <hip/hip_bf16.h>
#include <math.h>

typedef __attribute__((ext_vector_type(16))) _Float16 v16h;
typedef __attribute__((ext_vector_type(8)))  _Float16 v8h;
typedef __attribute__((ext_vector_type(8)))  float    v8f;

#define PI_F          3.14159265358979323846f
#define TWO_PI_F      6.28318530717958647692f
#define INV_TWO_PI_F  0.15915494309189533577f

#define L    1024   // fine lattice
#define LC   512    // coarse lattice
#define FT   37     // fine tile rows/cols incl. halo (-2..+34 around 32-wide tile)
#define FP   40     // padded fine-tile LDS stride
#define CT   18     // coarse feature tile incl. halo 1 (16+2)
#define CP   19     // padded feature stride
#define HP   17     // padded hidden stride (gcd(17,32)=1 -> conflict free)
#define IMS  72     // im2col row stride in f16 (144B: 16B-aligned, 36-bank stride)

__device__ __forceinline__ float regularize(float t) {
    float r = t + PI_F;
    r -= floorf(r * INV_TWO_PI_F) * TWO_PI_F;
    return r - PI_F;
}

__global__ __launch_bounds__(256)
void blocker_fused_kernel(const float* __restrict__ fine,
                          const float* __restrict__ w1,
                          const float* __restrict__ b1,
                          const float* __restrict__ w2,
                          const float* __restrict__ b2,
                          float* __restrict__ out)
{
    __shared__ float    s_ux[FT * FP];
    __shared__ float    s_uy[FT * FP];
    __shared__ float    s_feat[4 * CT * CP];     // [ic][18][19]
    __shared__ _Float16 s_im2col[256 * IMS];     // [pixel][72], K 0..63 valid, 36.. zero
    __shared__ float    s_h[256 * HP];           // [pixel][17] hidden activations
    __shared__ _Float16 s_w1ph[16 * 64];         // padded weights: [oc][k], 0 for k>=36
    __shared__ float    s_b1[16];
    __shared__ float    s_w2[96];                // (6,16)
    __shared__ float    s_b2[6];

    const int tid = threadIdx.x;
    const int tx  = blockIdx.x;   // coarse tile x (0..31)
    const int ty  = blockIdx.y;   // coarse tile y (0..31)
    const int b   = blockIdx.z;   // batch (0..15)

    const float* ux = fine + (size_t)b * 2u * L * L;
    const float* uy = ux + (size_t)L * L;

    // ---- stage weights into LDS (padded f16 im2col-weight table) ----
    for (int idx = tid; idx < 16 * 64; idx += 256) {
        const int oc = idx >> 6;
        const int k  = idx & 63;
        s_w1ph[idx] = (k < 36) ? (_Float16)w1[oc * 36 + k] : (_Float16)0.0f;
    }
    if (tid < 16) s_b1[tid] = b1[tid];
    if (tid < 96) s_w2[tid] = w2[tid];
    if (tid < 6)  s_b2[tid] = b2[tid];

    // ---- load fine-field tile with halo (rows/cols -2..+34, periodic wrap) ----
    const int fy0 = ty * 32 - 2;
    const int fx0 = tx * 32 - 2;
    for (int idx = tid; idx < FT * FT; idx += 256) {
        const int r  = idx / FT, c = idx % FT;
        const int gr = (fy0 + r) & (L - 1);
        const int gc = (fx0 + c) & (L - 1);
        s_ux[r * FP + c] = ux[(size_t)gr * L + gc];
        s_uy[r * FP + c] = uy[(size_t)gr * L + gc];
    }
    __syncthreads();

    // ---- features: cos(plaq) at 4 parities, 18x18 coarse incl. halo ----
    for (int idx = tid; idx < 4 * CT * CT; idx += 256) {
        const int ch  = idx / (CT * CT);
        const int rem = idx % (CT * CT);
        const int j   = rem / CT, i = rem % CT;
        const int py  = ch & 1;          // ch1,ch3 -> row parity 1
        const int px  = ch >> 1;         // ch2,ch3 -> col parity 1
        const int r   = 2 * j + py;
        const int c   = 2 * i + px;
        const float p = s_ux[r * FP + c] + s_uy[(r + 1) * FP + c]
                      - s_ux[r * FP + c + 1] - s_uy[r * FP + c];
        s_feat[(ch * CT + j) * CP + i] = __cosf(p);
    }
    __syncthreads();

    // ---- im2col staging: one pixel row per thread, all k indices compile-time ----
    {
        const int p  = tid;
        const int py = p >> 4, px = p & 15;
        _Float16* row = &s_im2col[p * IMS];
        // zero the K padding region [32,64) (64B, 16B-aligned)
        {
            int4  zz = {0, 0, 0, 0};
            int4* zp = (int4*)(row + 32);
            zp[0] = zz; zp[1] = zz; zp[2] = zz; zp[3] = zz;
        }
#pragma unroll
        for (int k = 0; k < 36; ++k) {
            const int ic  = k / 9;        // compile-time
            const int tap = k % 9;
            const int dy  = tap / 3;
            const int dx  = tap % 3;
            row[k] = (_Float16)s_feat[(ic * CT + (py + dy)) * CP + (px + dx)];
        }
    }
    __syncthreads();

    // ---- 3x3 conv (4->16 ch) as implicit-GEMM WMMA: K = 36 zero-padded to 64 ----
    const int lane = tid & 31;
    const int wave = tid >> 5;
    const int half = lane >> 4;
    const int lrow = lane & 15;   // A row (pixel col) and B col (out channel)

    // B fragments: contiguous 16-f16 (32B) LDS loads, affine in half
    v16h bfrag[2];
#pragma unroll
    for (int blk = 0; blk < 2; ++blk)
        bfrag[blk] = *(const v16h*)&s_w1ph[lrow * 64 + 32 * blk + 16 * half];

#pragma unroll
    for (int t = 0; t < 2; ++t) {
        const int mtile = wave * 2 + t;   // coarse row of this 16-pixel M tile
        const _Float16* arow = &s_im2col[(mtile * 16 + lrow) * IMS];
        v8f acc = {};
#pragma unroll
        for (int blk = 0; blk < 2; ++blk) {
            // A fragment: two contiguous 8-f16 (16B) chunks in contiguous k order (same as B)
            const v8h lo = *(const v8h*)(arow + 32 * blk + 16 * half);        // K: 32b+16h+e
            const v8h hi = *(const v8h*)(arow + 32 * blk + 16 * half + 8);   // K: 32b+16h+8+e
            v16h afrag;
#pragma unroll
            for (int e = 0; e < 8; ++e) { afrag[e] = lo[e]; afrag[e + 8] = hi[e]; }
            acc = __builtin_amdgcn_wmma_f32_16x16x32_f16(
                      false, afrag, false, bfrag[blk], (short)0, acc, false, false);
        }
        // D layout: lane holds col N=lrow (out ch), VGPR r -> row M = r + 8*half
#pragma unroll
        for (int r = 0; r < 8; ++r) {
            const int pcol = r + 8 * half;          // coarse col within tile
            const int p    = mtile * 16 + pcol;     // pixel id 0..255
            const float hv = acc[r] + s_b1[lrow];
            s_h[p * HP + lrow] = hv > 0.0f ? hv : 0.0f;
        }
    }
    __syncthreads();

    // ---- per-pixel tail: 1x1 conv, softmax gates, paths, circular average ----
    {
        const int p  = tid;
        const int py = p >> 4, px = p & 15;

        float lg[6];
#pragma unroll
        for (int o = 0; o < 6; ++o) {
            float s = s_b2[o];
#pragma unroll
            for (int hh = 0; hh < 16; ++hh)
                s += s_w2[o * 16 + hh] * s_h[p * HP + hh];
            lg[o] = s;
        }

        float wx0, wx1, wx2, wy0, wy1, wy2;
        {
            const float m = fmaxf(lg[0], fmaxf(lg[1], lg[2]));
            const float e0 = __expf(lg[0] - m), e1 = __expf(lg[1] - m), e2 = __expf(lg[2] - m);
            const float inv = 1.0f / (e0 + e1 + e2);
            wx0 = e0 * inv; wx1 = e1 * inv; wx2 = e2 * inv;
        }
        {
            const float m = fmaxf(lg[3], fmaxf(lg[4], lg[5]));
            const float e0 = __expf(lg[3] - m), e1 = __expf(lg[4] - m), e2 = __expf(lg[5] - m);
            const float inv = 1.0f / (e0 + e1 + e2);
            wy0 = e0 * inv; wy1 = e1 * inv; wy2 = e2 * inv;
        }

        const int fy = 2 * py + 2;   // local fine coords (halo offset +2)
        const int fx = 2 * px + 2;
#define UXS(r, c) s_ux[(r) * FP + (c)]
#define UYS(r, c) s_uy[(r) * FP + (c)]
        // x paths (sampled at even-even fine sites)
        const float xp0 = UXS(fy, fx) + UXS(fy + 1, fx);
        const float xp1 = UYS(fy, fx) + UXS(fy, fx + 1) + UXS(fy + 1, fx + 1) - UYS(fy + 2, fx);
        const float xp2 = -UYS(fy, fx - 1) + UXS(fy, fx - 1) + UXS(fy + 1, fx - 1) + UYS(fy + 2, fx - 1);
        // y paths
        const float yp0 = UYS(fy, fx) + UYS(fy, fx + 1);
        const float yp1 = UXS(fy, fx) + UYS(fy + 1, fx) + UYS(fy + 1, fx + 1) - UXS(fy, fx + 2);
        const float yp2 = -UXS(fy - 1, fx) + UYS(fy - 1, fx) + UYS(fy - 1, fx + 1) + UXS(fy - 1, fx + 2);
#undef UXS
#undef UYS
        // regularize before sin/cos is a mathematical no-op (2*pi periodic)
        const float sx = wx0 * __sinf(xp0) + wx1 * __sinf(xp1) + wx2 * __sinf(xp2);
        const float cx = wx0 * __cosf(xp0) + wx1 * __cosf(xp1) + wx2 * __cosf(xp2);
        const float sy = wy0 * __sinf(yp0) + wy1 * __sinf(yp1) + wy2 * __sinf(yp2);
        const float cy = wy0 * __cosf(yp0) + wy1 * __cosf(yp1) + wy2 * __cosf(yp2);

        const float ox = regularize(atan2f(sx, cx));
        const float oy = regularize(atan2f(sy, cy));

        const int gy = ty * 16 + py;
        const int gx = tx * 16 + px;
        const size_t obase = (size_t)b * 2u * LC * LC;
        out[obase + (size_t)gy * LC + gx]                   = ox;
        out[obase + (size_t)LC * LC + (size_t)gy * LC + gx] = oy;
    }
}

extern "C" void kernel_launch(void* const* d_in, const int* in_sizes, int n_in,
                              void* d_out, int out_size, void* d_ws, size_t ws_size,
                              hipStream_t stream) {
    const float* fine = (const float*)d_in[0];   // (16,2,1024,1024) f32
    const float* w1   = (const float*)d_in[1];   // (16,4,3,3)
    const float* b1   = (const float*)d_in[2];   // (16,)
    const float* w2   = (const float*)d_in[3];   // (6,16,1,1)
    const float* b2   = (const float*)d_in[4];   // (6,)
    float* out = (float*)d_out;                  // (16,2,512,512) f32

    dim3 grid(32, 32, 16);   // 16x16 coarse tile per block, per batch image
    blocker_fused_kernel<<<grid, 256, 0, stream>>>(fine, w1, b1, w2, b2, out);
}
